// BidirectionalAttention_51874615001166
// MI455X (gfx1250) — matrix operand order repaired and checked by reading, not compiled, not run on hardware
//
#include <hip/hip_runtime.h>

#define DEV __device__ __forceinline__

typedef __bf16 bf16_t;
typedef __attribute__((ext_vector_type(16))) __bf16        v16bf;
typedef __attribute__((ext_vector_type(8)))  float         v8f;
typedef __attribute__((ext_vector_type(4)))  unsigned int  v4u;
typedef __attribute__((ext_vector_type(8)))  unsigned int  v8u;

static constexpr int BATCH = 16;
static constexpr int L     = 1024;   // L1 == L2
static constexpr int D     = 512;

// ---- fp32 -> bf16 (round to nearest even) ------------------------------
DEV unsigned short f2bf_bits(float x) {
    union { float f; unsigned int u; } v; v.f = x;
    unsigned int r = v.u + 0x7FFFu + ((v.u >> 16) & 1u);
    return (unsigned short)(r >> 16);
}
DEV bf16_t f2bf(float x) {
    unsigned short h = f2bf_bits(x);
    return __builtin_bit_cast(bf16_t, h);
}

// ---- WMMA fragment loaders (wave32, v_wmma_f32_16x16x32_bf16) ----------
// A (16x32, 16-bit): lane<16 holds row lane, K = {0..7, 16..23};
//                    lane>=16 holds row lane-16, K = {8..15, 24..31}.
DEV v16bf load_fragA(const bf16_t* tile, int ld, int lrow, int hf, int kb) {
    const bf16_t* p = tile + (size_t)lrow * ld + kb + hf * 8;
    v4u lo = *(const v4u*)p;          // K +0..7
    v4u hi = *(const v4u*)(p + 16);   // K +16..23
    v8u r;
    r[0] = lo[0]; r[1] = lo[1]; r[2] = lo[2]; r[3] = lo[3];
    r[4] = hi[0]; r[5] = hi[1]; r[6] = hi[2]; r[7] = hi[3];
    return __builtin_bit_cast(v16bf, r);
}
// B (32x16, 16-bit): lane<16 holds col lane, K = {0..7, 16..23};
//                    lane>=16 holds col lane-16, K = {8..15, 24..31}.
// From an LDS tile laid out [cols][32 k] bf16 (row pitch 64 B).
DEV v16bf load_fragB_lds(const bf16_t* tile, int lcol, int hf) {
    const bf16_t* p = tile + lcol * 32 + hf * 8;
    v4u lo = *(const v4u*)p;          // 16 B -> ds_load_b128
    v4u hi = *(const v4u*)(p + 16);
    v8u r;
    r[0] = lo[0]; r[1] = lo[1]; r[2] = lo[2]; r[3] = lo[3];
    r[4] = hi[0]; r[5] = hi[1]; r[6] = hi[2]; r[7] = hi[3];
    return __builtin_bit_cast(v16bf, r);
}

// ---- Tensor Data Mover: 2D tile (rows x 32 bf16) global -> LDS ---------
// D# per CDNA5 ISA ch.8: group0 = {count/lds_addr/global_addr/type},
// group1 = {data_size, tensor dims, tile dims, dim0 stride}. 2-group form
// (tile is 2D), data_size=1 (2 bytes), tile_dim0 = 32 elements (64 B).
// The LDS pointer is passed as an asm operand so the buffer escapes and
// the "memory" clobber makes the TDM visible as a writer of the tile.
typedef __attribute__((ext_vector_type(4))) unsigned int u32x4;
typedef __attribute__((ext_vector_type(8))) unsigned int u32x8;

DEV void tdm_load_2d(bf16_t* lds_ptr, const void* gptr,
                     unsigned tile_rows, unsigned row_stride_elems) {
    // Flat address of an LDS object: addr[31:0] is the LDS byte offset.
    unsigned lds_byte_off = (unsigned)(unsigned long long)(size_t)lds_ptr;
    unsigned long long ga = (unsigned long long)gptr;
    u32x4 g0;
    g0[0] = 1u;                                   // count=1, user descriptor
    g0[1] = lds_byte_off;                         // LDS destination (bytes)
    g0[2] = (unsigned)ga;                         // global_addr[31:0]
    g0[3] = (unsigned)(ga >> 32) | (2u << 30);    // addr[56:32] | type=2
    const unsigned td0 = 0x40000000u;             // huge tensor dims: no OOB clip
    const unsigned td1 = 0x40000000u;
    u32x8 g1;
    g1[0] = 0x00010000u;                          // wg_mask=0, data_size=1 (2B)
    g1[1] = (td0 & 0xFFFFu) << 16;                // tensor_dim0[15:0]
    g1[2] = (td0 >> 16) | ((td1 & 0xFFFFu) << 16);// tensor_dim0 hi | dim1 lo
    g1[3] = (td1 >> 16) | (32u << 16);            // tensor_dim1 hi | tile_dim0=32
    g1[4] = tile_rows & 0xFFFFu;                  // tile_dim1 | tile_dim2=0
    g1[5] = row_stride_elems;                     // tensor_dim0_stride[31:0]
    g1[6] = 0u;
    g1[7] = 0u;
    asm volatile("tensor_load_to_lds %0, %1"
                 :: "s"(g0), "s"(g1), "r"(lds_ptr) : "memory");
}

// ---- K0: fp32 -> bf16 copies (row-major and transposed) ----------------
__global__ void k_convert(const float* __restrict__ v1, const float* __restrict__ v2,
                          bf16_t* __restrict__ v1b, bf16_t* __restrict__ v2b,
                          bf16_t* __restrict__ v1t, bf16_t* __restrict__ v2t) {
    size_t idx = (size_t)blockIdx.x * blockDim.x + threadIdx.x;
    int d = (int)(idx % D);
    size_t t = idx / D;
    int l = (int)(t % L);
    int b = (int)(t / L);
    bf16_t a = f2bf(v1[idx]);
    bf16_t c = f2bf(v2[idx]);
    v1b[idx] = a;
    v2b[idx] = c;
    size_t tidx = ((size_t)b * D + d) * L + l;
    v1t[tidx] = a;
    v2t[tidx] = c;
}

// ---- K1: sim[b,i,j] = sum_d v1[b,i,d]*v2[b,j,d] ------------------------
// 256 threads (8 waves). WG tile 128x128; wave strip 16x128.
// B tile chunk (128 j x 32 k bf16, 8 KB) staged by TDM, double-buffered.
__global__ void __launch_bounds__(256)
k_gemm_sim(const bf16_t* __restrict__ v1b,
           const bf16_t* __restrict__ v2b,
           float* __restrict__ sim) {
    __shared__ bf16_t Bs[2][128 * 32];            // 2 x 8 KB
    const int wave = threadIdx.x >> 5, lane = threadIdx.x & 31;
    const int lrow = lane & 15, hf = lane >> 4;
    const int b = blockIdx.z;
    const int i_base = blockIdx.x * 128 + wave * 16;
    const int j_base = blockIdx.y * 128;
    const bf16_t* Atile = v1b + ((size_t)b * L + i_base) * D;
    const bf16_t* Bglob = v2b + ((size_t)b * L + j_base) * D;   // 128 rows, pitch D

    if (threadIdx.x == 0)
        tdm_load_2d(Bs[0], Bglob, 128, D);        // chunk 0 -> buffer 0

    v8f acc[8] = {};
    for (int kk = 0; kk < D; kk += 32) {
        const int c   = kk >> 5;
        const int cur = c & 1;
        if (threadIdx.x == 0) {
            if (kk + 32 < D) {
                tdm_load_2d(Bs[cur ^ 1], Bglob + (kk + 32), 128, D);
                __builtin_amdgcn_s_wait_tensorcnt(1);   // chunk c landed
            } else {
                __builtin_amdgcn_s_wait_tensorcnt(0);
            }
        }
        __syncthreads();                          // chunk c visible to all waves

        if (kk + 32 < D)
            __builtin_prefetch(Atile + (size_t)lrow * D + kk + 32, 0, 3);
        v16bf a = load_fragA(Atile, D, lrow, hf, kk);
        const bf16_t* Bt = Bs[cur];
#pragma unroll
        for (int t = 0; t < 8; ++t) {
            v16bf bm = load_fragB_lds(Bt + t * 16 * 32, lrow, hf);
            acc[t] = __builtin_amdgcn_wmma_f32_16x16x32_bf16(
                false, a, false, bm, (short)0, acc[t], false, false);
        }
        __syncthreads();                          // done reading buf before reuse
    }

    float* srow = sim + ((size_t)b * L + i_base) * L + j_base;
#pragma unroll
    for (int t = 0; t < 8; ++t)
#pragma unroll
        for (int r = 0; r < 8; ++r) {
            int row = r + hf * 8;
            srow[(size_t)row * L + t * 16 + lrow] = acc[t][r];
        }
}

// ---- K2: row stats (softmax over axis 2, mask = v2_mask[j]) ------------
__global__ void k_rowstats(const float* __restrict__ sim,
                           const unsigned char* __restrict__ v2m,
                           float* __restrict__ mx, float* __restrict__ sm) {
    int wave = threadIdx.x >> 5, lane = threadIdx.x & 31;
    int w = blockIdx.x * (blockDim.x >> 5) + wave;   // 0 .. B*L-1
    int b = w >> 10, i = w & (L - 1);
    const float* row = sim + ((size_t)b * L + i) * L;
    const unsigned char* msk = v2m + b * L;
    float m = -INFINITY, s = 0.f;
    for (int j = lane; j < L; j += 32) {
        if (!msk[j]) {
            float x = row[j];
            float M = fmaxf(m, x);
            s = s * __expf(m - M) + __expf(x - M);
            m = M;
        }
    }
#pragma unroll
    for (int off = 16; off > 0; off >>= 1) {
        float mo = __shfl_xor(m, off, 32);
        float so = __shfl_xor(s, off, 32);
        float M = fmaxf(m, mo);
        if (M == -INFINITY) {
            s = s + so;                           // both empty: avoid NaN
        } else {
            s = s * __expf(m - M) + so * __expf(mo - M);
        }
        m = M;
    }
    if (lane == 0) { mx[w] = m; sm[w] = s; }
}

// ---- K3: column stats (softmax over axis 1, mask = v1_mask[i]) ---------
__global__ void k_colstats(const float* __restrict__ sim,
                           const unsigned char* __restrict__ v1m,
                           float* __restrict__ mx, float* __restrict__ sm) {
    int t = blockIdx.x * blockDim.x + threadIdx.x;   // 0 .. B*L-1
    int b = t >> 10, j = t & (L - 1);
    const unsigned char* msk = v1m + b * L;
    const float* base = sim + (size_t)b * L * L + j;
    float m = -INFINITY, s = 0.f;
    for (int i = 0; i < L; ++i) {
        if (!msk[i]) {
            float x = base[(size_t)i * L];
            float M = fmaxf(m, x);
            s = s * __expf(m - M) + __expf(x - M);
            m = M;
        }
    }
    mx[t] = m;
    sm[t] = s;
}

// ---- K4/K5: out[b,r,d] = sum_k P[r,k] * V[k,d] -------------------------
// 256 threads (8 waves). WG tile: 128 rows x 64 d-cols; wave strip 16x64.
// P built on the fly from sim (strides allow transposed read for dir 2).
// V^T tile (64 d x 32 k bf16, 4 KB) staged by TDM, double-buffered.
__global__ void __launch_bounds__(256)
k_attend(const float* __restrict__ sim, int srow_, int sk_,
         const float* __restrict__ mx, const float* __restrict__ sm,
         const unsigned char* __restrict__ innerM,
         const unsigned char* __restrict__ outerM,
         const bf16_t* __restrict__ Vt,
         float* __restrict__ out) {
    __shared__ bf16_t Vs[2][64 * 32];             // 2 x 4 KB
    const int wave = threadIdx.x >> 5, lane = threadIdx.x & 31;
    const int lrow = lane & 15, hf = lane >> 4;
    const int b = blockIdx.z;
    const int r_base = blockIdx.x * 128 + wave * 16;
    const int d_base = blockIdx.y * 64;

    const int gr = r_base + lrow;                 // this lane's P row
    const float m   = mx[b * L + gr];
    const float inv = 1.0f / sm[b * L + gr];
    const float* prow = sim + (size_t)b * L * L + (size_t)gr * srow_;
    const unsigned char* im = innerM + b * L;
    const bf16_t* Vglob = Vt + ((size_t)b * D + d_base) * L;    // 64 rows, pitch L

    if (threadIdx.x == 0)
        tdm_load_2d(Vs[0], Vglob, 64, L);         // chunk 0 -> buffer 0

    v8f acc[4] = {};
    for (int kb = 0; kb < L; kb += 32) {
        const int c   = kb >> 5;
        const int cur = c & 1;
        if (threadIdx.x == 0) {
            if (kb + 32 < L) {
                tdm_load_2d(Vs[cur ^ 1], Vglob + (kb + 32), 64, L);
                __builtin_amdgcn_s_wait_tensorcnt(1);
            } else {
                __builtin_amdgcn_s_wait_tensorcnt(0);
            }
        }
        __syncthreads();

        // Build A (probability) fragment in registers.
        v8u aw;
#pragma unroll
        for (int u = 0; u < 4; ++u) {
            int k0 = kb + hf * 8 + 2 * u;
            float x0 = prow[(size_t)k0 * sk_];
            float x1 = prow[(size_t)(k0 + 1) * sk_];
            float p0 = im[k0]     ? 0.f : __expf(x0 - m) * inv;
            float p1 = im[k0 + 1] ? 0.f : __expf(x1 - m) * inv;
            aw[u] = (unsigned int)f2bf_bits(p0) | ((unsigned int)f2bf_bits(p1) << 16);
        }
#pragma unroll
        for (int u = 0; u < 4; ++u) {
            int k0 = kb + 16 + hf * 8 + 2 * u;
            float x0 = prow[(size_t)k0 * sk_];
            float x1 = prow[(size_t)(k0 + 1) * sk_];
            float p0 = im[k0]     ? 0.f : __expf(x0 - m) * inv;
            float p1 = im[k0 + 1] ? 0.f : __expf(x1 - m) * inv;
            aw[4 + u] = (unsigned int)f2bf_bits(p0) | ((unsigned int)f2bf_bits(p1) << 16);
        }
        v16bf a = __builtin_bit_cast(v16bf, aw);
        const bf16_t* Bt = Vs[cur];
#pragma unroll
        for (int t = 0; t < 4; ++t) {
            v16bf bm = load_fragB_lds(Bt + t * 16 * 32, lrow, hf);
            acc[t] = __builtin_amdgcn_wmma_f32_16x16x32_bf16(
                false, a, false, bm, (short)0, acc[t], false, false);
        }
        __syncthreads();
    }

    const unsigned char* om = outerM + b * L;
    float* orow = out + ((size_t)b * L + r_base) * D + d_base;
#pragma unroll
    for (int t = 0; t < 4; ++t)
#pragma unroll
        for (int r = 0; r < 8; ++r) {
            int row = r + hf * 8;
            float v = om[r_base + row] ? 0.f : acc[t][r];
            orow[(size_t)row * D + t * 16 + lrow] = v;
        }
}

extern "C" void kernel_launch(void* const* d_in, const int* in_sizes, int n_in,
                              void* d_out, int out_size, void* d_ws, size_t ws_size,
                              hipStream_t stream) {
    const float*         v1  = (const float*)d_in[0];
    const unsigned char* v1m = (const unsigned char*)d_in[1];
    const float*         v2  = (const float*)d_in[2];
    const unsigned char* v2m = (const unsigned char*)d_in[3];

    const size_t NV = (size_t)BATCH * L * D;   // 8,388,608 elements
    float* out1 = (float*)d_out;               // attended_v1 [B,L1,D]
    float* out2 = out1 + NV;                   // attended_v2 [B,L2,D]

    char* ws = (char*)d_ws;
    bf16_t* v1b = (bf16_t*)ws; ws += NV * 2;
    bf16_t* v2b = (bf16_t*)ws; ws += NV * 2;
    bf16_t* v1t = (bf16_t*)ws; ws += NV * 2;   // [B, D, L1]
    bf16_t* v2t = (bf16_t*)ws; ws += NV * 2;   // [B, D, L2]
    float*  sim = (float*)ws;  ws += (size_t)BATCH * L * L * 4;   // 67 MB, fits L2
    float*  mx1 = (float*)ws;  ws += (size_t)BATCH * L * 4;
    float*  sm1 = (float*)ws;  ws += (size_t)BATCH * L * 4;
    float*  mx2 = (float*)ws;  ws += (size_t)BATCH * L * 4;
    float*  sm2 = (float*)ws;  ws += (size_t)BATCH * L * 4;

    k_convert<<<(unsigned)(NV / 256), 256, 0, stream>>>(v1, v2, v1b, v2b, v1t, v2t);

    k_gemm_sim<<<dim3(L / 128, L / 128, BATCH), 256, 0, stream>>>(v1b, v2b, sim);

    k_rowstats<<<(BATCH * L) / 8, 256, 0, stream>>>(sim, v2m, mx1, sm1);
    k_colstats<<<(BATCH * L) / 256, 256, 0, stream>>>(sim, v1m, mx2, sm2);

    // attended_v1: rows = i (softmax over j), inner mask v2, V = v2^T
    k_attend<<<dim3(L / 128, D / 64, BATCH), 256, 0, stream>>>(
        sim, L, 1, mx1, sm1, v2m, v1m, v2t, out1);
    // attended_v2: rows = j (softmax over i, read sim transposed), V = v1^T
    k_attend<<<dim3(L / 128, D / 64, BATCH), 256, 0, stream>>>(
        sim, 1, L, mx2, sm2, v1m, v2m, v1t, out2);
}
